// PointNetPPEncoder_41205916238102
// MI455X (gfx1250) — compile-verified, matrix operand order checked by reading, not run on hardware
//
#include <hip/hip_runtime.h>
#include <hip/hip_bf16.h>

typedef __attribute__((ext_vector_type(16))) _Float16 v16h;
typedef __attribute__((ext_vector_type(8)))  _Float16 v8h;
typedef __attribute__((ext_vector_type(8)))  float    v8f;

#define NB 4        // batch
#define NIN 8192    // initial points
#define NS 32       // nsample

// ---------------------------------------------------------------------------
// Stage 0 init: xyz (B,6,8192) -> xyz0 (B,N,3), pts0 (B,N,6), and the first
// two outputs l_xyz[0]=(B,3,N), l_points[0]=(B,6,N).
// ---------------------------------------------------------------------------
__global__ void k_init(const float* __restrict__ xyz,
                       float* __restrict__ xyz0, float* __restrict__ pts0,
                       float* __restrict__ out_lxyz0, float* __restrict__ out_lpts0)
{
    int gi = blockIdx.x * blockDim.x + threadIdx.x;
    if (gi >= NB * NIN) return;
    int b = gi >> 13, n = gi & (NIN - 1);
    #pragma unroll
    for (int c = 0; c < 6; ++c) {
        float v = xyz[((size_t)b * 6 + c) * NIN + n];
        pts0[((size_t)b * NIN + n) * 6 + c] = v;
        out_lpts0[((size_t)b * 6 + c) * NIN + n] = v;
        if (c < 3) {
            xyz0[((size_t)b * NIN + n) * 3 + c] = v;
            out_lxyz0[((size_t)b * 3 + c) * NIN + n] = v;
        }
    }
}

// ---------------------------------------------------------------------------
// Farthest point sampling: one workgroup per batch, dist[] lives in LDS.
// Sequential npoint iterations, argmax via LDS tree reduce. Records `far`
// BEFORE update (matches the jax scan, first index is 0).
// ---------------------------------------------------------------------------
#define FPS_T 256
__global__ void k_fps(const float* __restrict__ xyz,  // [B][N][3]
                      int* __restrict__ out_idx,      // [B][npoint]
                      int N, int npoint)
{
    int b = blockIdx.x;
    const float* P = xyz + (size_t)b * N * 3;
    extern __shared__ float smem[];
    float* dist = smem;                 // N floats
    float* rmax = smem + N;             // FPS_T floats
    int*   rarg = (int*)(rmax + FPS_T); // FPS_T ints
    __shared__ int s_far;
    int t = threadIdx.x;
    for (int i = t; i < N; i += FPS_T) dist[i] = 1e10f;
    if (t == 0) s_far = 0;
    __syncthreads();

    for (int it = 0; it < npoint; ++it) {
        int far = s_far;
        if (t == 0) out_idx[b * npoint + it] = far;
        float cx = P[far * 3 + 0], cy = P[far * 3 + 1], cz = P[far * 3 + 2];
        float bmax = -1.0f; int barg = 0;
        for (int i = t; i < N; i += FPS_T) {
            float dx = P[i * 3 + 0] - cx;
            float dy = P[i * 3 + 1] - cy;
            float dz = P[i * 3 + 2] - cz;
            float d = dx * dx + dy * dy + dz * dz;
            float nd = fminf(dist[i], d);
            dist[i] = nd;
            if (nd > bmax) { bmax = nd; barg = i; }
        }
        rmax[t] = bmax; rarg[t] = barg;
        __syncthreads();
        for (int s = FPS_T / 2; s > 0; s >>= 1) {
            if (t < s) {
                if (rmax[t + s] > rmax[t] ||
                    (rmax[t + s] == rmax[t] && rarg[t + s] < rarg[t])) {
                    rmax[t] = rmax[t + s]; rarg[t] = rarg[t + s];
                }
            }
            __syncthreads();
        }
        if (t == 0) s_far = rarg[0];
        __syncthreads();
    }
}

// ---------------------------------------------------------------------------
// Gather new_xyz (B,S,3) from fps indices; also emit l_xyz[i+1] = (B,3,S).
// ---------------------------------------------------------------------------
__global__ void k_newxyz(const float* __restrict__ xyz, const int* __restrict__ fidx,
                         float* __restrict__ new_xyz, float* __restrict__ out_lxyz,
                         int N, int S)
{
    int gi = blockIdx.x * blockDim.x + threadIdx.x;
    if (gi >= NB * S) return;
    int b = gi / S, s = gi % S;
    int j = fidx[b * S + s];
    #pragma unroll
    for (int c = 0; c < 3; ++c) {
        float v = xyz[((size_t)b * N + j) * 3 + c];
        new_xyz[((size_t)b * S + s) * 3 + c] = v;
        out_lxyz[((size_t)b * 3 + c) * S + s] = v;
    }
}

// ---------------------------------------------------------------------------
// Ball query: one thread per (b,s); first NS point indices with d2<=r2, padded
// with the first hit (matches top_k over -gidx selecting smallest indices).
// ---------------------------------------------------------------------------
__global__ void k_ball(const float* __restrict__ xyz, const float* __restrict__ new_xyz,
                       int* __restrict__ idx, int N, int S, float r2)
{
    int gi = blockIdx.x * blockDim.x + threadIdx.x;
    if (gi >= NB * S) return;
    int b = gi / S, s = gi % S;
    const float* P = xyz + (size_t)b * N * 3;
    const float* q = new_xyz + ((size_t)b * S + s) * 3;
    float cx = q[0], cy = q[1], cz = q[2];
    int* o = idx + ((size_t)b * S + s) * NS;
    int cnt = 0, first = 0;
    for (int i = 0; i < N && cnt < NS; ++i) {
        float dx = P[i * 3 + 0] - cx;
        float dy = P[i * 3 + 1] - cy;
        float dz = P[i * 3 + 2] - cz;
        float d = dx * dx + dy * dy + dz * dz;
        if (d <= r2) { if (cnt == 0) first = i; o[cnt++] = i; }
    }
    for (; cnt < NS; ++cnt) o[cnt] = first;
}

// ---------------------------------------------------------------------------
// Grouping: build GEMM input X[m][Cpad] (half), m = (b*S+s)*32+k.
// c<3: grouped_xyz (centered); 3..3+C: gathered features; rest zero-pad.
// ---------------------------------------------------------------------------
__global__ void k_group(const float* __restrict__ xyz, const float* __restrict__ pts,
                        const float* __restrict__ new_xyz, const int* __restrict__ idx,
                        _Float16* __restrict__ X, int N, int S, int C, int Cpad)
{
    int m = blockIdx.x * blockDim.x + threadIdx.x;
    int M = NB * S * NS;
    if (m >= M) return;
    int k = m & (NS - 1);
    int bs = m / NS;
    int s = bs % S, b = bs / S;
    int j = idx[((size_t)b * S + s) * NS + k];
    const float* p = xyz + ((size_t)b * N + j) * 3;
    const float* q = new_xyz + ((size_t)b * S + s) * 3;
    _Float16* o = X + (size_t)m * Cpad;
    o[0] = (_Float16)(p[0] - q[0]);
    o[1] = (_Float16)(p[1] - q[1]);
    o[2] = (_Float16)(p[2] - q[2]);
    const float* f = pts + ((size_t)b * N + j) * C;
    for (int c = 0; c < C; ++c) o[3 + c] = (_Float16)f[c];
    for (int c = 3 + C; c < Cpad; ++c) o[c] = (_Float16)0.0f;
}

// ---------------------------------------------------------------------------
// Weight f32 -> half, K padded with zeros to Cpad.
// ---------------------------------------------------------------------------
__global__ void k_w2h(const float* __restrict__ W, _Float16* __restrict__ Wh,
                      int Cout, int Cin, int Cpad)
{
    int gi = blockIdx.x * blockDim.x + threadIdx.x;
    if (gi >= Cout * Cpad) return;
    int c = gi % Cpad, o = gi / Cpad;
    Wh[gi] = (c < Cin) ? (_Float16)W[o * Cin + c] : (_Float16)0.0f;
}

// ---------------------------------------------------------------------------
// WMMA GEMM: Y[Cout][M] = Wh(Cout x Cpad, half) * X(M x Cpad, half)^T + bias.
// 4 waves/block; each wave computes a 2x2 block of 16x16 tiles (32 output
// channels x 32 columns) so each fragment load feeds two WMMAs:
// 8 b128 loads -> 4 v_wmma_f32_16x16x32_f16 per K step.
// Per the ISA 16-bit fragment layout each lane needs K = kb..kb+7 and
// kb+16..kb+23 (kb = 0|8 by lane half) = two contiguous 16-byte loads since
// K is the fastest-varying dim of both Wh and X.
// ---------------------------------------------------------------------------
#define GEMM_WAVES 4
__global__ void k_gemm(const _Float16* __restrict__ Wh,  // [Cout][Cpad]
                       const _Float16* __restrict__ X,   // [M][Cpad]
                       const float* __restrict__ bias,   // [Cout]
                       float* __restrict__ Y,            // [Cout][M]
                       int Cpad, int M)
{
    int lane = threadIdx.x & 31;
    int wave = threadIdx.x >> 5;
    int m0 = (blockIdx.x * GEMM_WAVES + wave) * 32;   // 2 column tiles
    int o0 = blockIdx.y * 32;                         // 2 channel tiles
    int r  = lane & 15;
    int kb = (lane >> 4) << 3;   // 0 or 8

    const _Float16* aR0 = Wh + (size_t)(o0 + r) * Cpad + kb;
    const _Float16* aR1 = Wh + (size_t)(o0 + 16 + r) * Cpad + kb;
    const _Float16* bR0 = X  + (size_t)(m0 + r) * Cpad + kb;
    const _Float16* bR1 = X  + (size_t)(m0 + 16 + r) * Cpad + kb;

    v8f acc00 = {}, acc01 = {}, acc10 = {}, acc11 = {};
    for (int k = 0; k < Cpad; k += 32) {
        __builtin_prefetch((const void*)(bR0 + k + 64), 0, 0);
        __builtin_prefetch((const void*)(bR1 + k + 64), 0, 0);
        v8h a0l = *(const v8h*)(aR0 + k);
        v8h a0h = *(const v8h*)(aR0 + k + 16);
        v8h a1l = *(const v8h*)(aR1 + k);
        v8h a1h = *(const v8h*)(aR1 + k + 16);
        v8h b0l = *(const v8h*)(bR0 + k);
        v8h b0h = *(const v8h*)(bR0 + k + 16);
        v8h b1l = *(const v8h*)(bR1 + k);
        v8h b1h = *(const v8h*)(bR1 + k + 16);
        v16h a0 = __builtin_shufflevector(a0l, a0h, 0,1,2,3,4,5,6,7,8,9,10,11,12,13,14,15);
        v16h a1 = __builtin_shufflevector(a1l, a1h, 0,1,2,3,4,5,6,7,8,9,10,11,12,13,14,15);
        v16h b0 = __builtin_shufflevector(b0l, b0h, 0,1,2,3,4,5,6,7,8,9,10,11,12,13,14,15);
        v16h b1 = __builtin_shufflevector(b1l, b1h, 0,1,2,3,4,5,6,7,8,9,10,11,12,13,14,15);
        acc00 = __builtin_amdgcn_wmma_f32_16x16x32_f16(false, a0, false, b0, (short)0, acc00, false, false);
        acc01 = __builtin_amdgcn_wmma_f32_16x16x32_f16(false, a0, false, b1, (short)0, acc01, false, false);
        acc10 = __builtin_amdgcn_wmma_f32_16x16x32_f16(false, a1, false, b0, (short)0, acc10, false, false);
        acc11 = __builtin_amdgcn_wmma_f32_16x16x32_f16(false, a1, false, b1, (short)0, acc11, false, false);
    }

    // D layout: float j of lane L holds (row = j + 8*(L>>4), col = L&15).
    int rsel = (lane >> 4) << 3;
    int c0 = m0 + (lane & 15);
    #pragma unroll
    for (int j = 0; j < 8; ++j) {
        int oA = o0 + rsel + j;
        int oB = o0 + 16 + rsel + j;
        float bA = bias[oA], bB = bias[oB];
        Y[(size_t)oA * M + c0]      = acc00[j] + bA;
        Y[(size_t)oA * M + c0 + 16] = acc01[j] + bA;
        Y[(size_t)oB * M + c0]      = acc10[j] + bB;
        Y[(size_t)oB * M + c0 + 16] = acc11[j] + bB;
    }
}

// ---------------------------------------------------------------------------
// BN stats: one block per channel, exact mean / inv-std over all M columns.
// ---------------------------------------------------------------------------
__global__ void k_stats(const float* __restrict__ Y, float* __restrict__ mean,
                        float* __restrict__ invstd, int M)
{
    int c = blockIdx.x;
    const float* y = Y + (size_t)c * M;
    float s = 0.0f, ss = 0.0f;
    for (int i = threadIdx.x; i < M; i += blockDim.x) {
        float v = y[i]; s += v; ss += v * v;
    }
    __shared__ float sm[256], sq[256];
    sm[threadIdx.x] = s; sq[threadIdx.x] = ss;
    __syncthreads();
    for (int st = 128; st > 0; st >>= 1) {
        if (threadIdx.x < st) {
            sm[threadIdx.x] += sm[threadIdx.x + st];
            sq[threadIdx.x] += sq[threadIdx.x + st];
        }
        __syncthreads();
    }
    if (threadIdx.x == 0) {
        float mu  = sm[0] / (float)M;
        float var = sq[0] / (float)M - mu * mu;
        mean[c] = mu;
        invstd[c] = rsqrtf(var + 1e-5f);
    }
}

// ---------------------------------------------------------------------------
// BN + ReLU, repacked to half as next layer's X[m][Cout].
// ---------------------------------------------------------------------------
__global__ void k_bnrelu(const float* __restrict__ Y, const float* __restrict__ mean,
                         const float* __restrict__ invstd, const float* __restrict__ g,
                         const float* __restrict__ be, _Float16* __restrict__ Xn,
                         int Cout, int M)
{
    int m = blockIdx.x * blockDim.x + threadIdx.x;
    if (m >= M) return;
    _Float16* o = Xn + (size_t)m * Cout;
    for (int c = 0; c < Cout; ++c) {
        float v = Y[(size_t)c * M + m];
        v = g[c] * (v - mean[c]) * invstd[c] + be[c];
        o[c] = (_Float16)fmaxf(v, 0.0f);
    }
}

// ---------------------------------------------------------------------------
// Last layer: BN + ReLU fused with max over nsample. Writes l_points[i+1]
// (B,C,S) to d_out and (B,S,C) features for the next stage.
// ---------------------------------------------------------------------------
__global__ void k_maxpool(const float* __restrict__ Y, const float* __restrict__ mean,
                          const float* __restrict__ invstd, const float* __restrict__ g,
                          const float* __restrict__ be, float* __restrict__ out_feat,
                          float* __restrict__ pts_next, int S, int C)
{
    int gi = blockIdx.x * blockDim.x + threadIdx.x;
    if (gi >= NB * S * C) return;
    int c = gi % C;
    int bs = gi / C;
    int s = bs % S, b = bs / S;
    size_t M = (size_t)NB * S * NS;
    const float* y = Y + (size_t)c * M + ((size_t)b * S + s) * NS;
    float gc = g[c], mc = mean[c], ic = invstd[c], bc = be[c];
    float mx = -1e30f;
    #pragma unroll
    for (int k = 0; k < NS; ++k) {
        float v = gc * (y[k] - mc) * ic + bc;
        v = fmaxf(v, 0.0f);
        mx = fmaxf(mx, v);
    }
    out_feat[((size_t)b * C + c) * S + s] = mx;
    pts_next[((size_t)b * S + s) * C + c] = mx;
}

// ---------------------------------------------------------------------------
// Host orchestration
// ---------------------------------------------------------------------------
static const int   H_NPOINTS[4] = {2048, 1024, 512, 256};
static const float H_RADII[4]   = {0.05f, 0.1f, 0.2f, 0.3f};
static const int   H_MLPS[4][3] = {{32,32,64},{64,64,128},{128,128,256},{256,256,512}};

extern "C" void kernel_launch(void* const* d_in, const int* in_sizes, int n_in,
                              void* d_out, int out_size, void* d_ws, size_t ws_size,
                              hipStream_t stream)
{
    (void)in_sizes; (void)n_in; (void)out_size; (void)ws_size;
    const float* xyz_in = (const float*)d_in[0];
    float* out = (float*)d_out;

    // ---- output offsets: l_xyz[0..4] then l_points[0..4]
    size_t o_lxyz[5], o_lpts[5], off = 0;
    o_lxyz[0] = off; off += (size_t)NB * 3 * NIN;
    for (int i = 0; i < 4; ++i) { o_lxyz[i + 1] = off; off += (size_t)NB * 3 * H_NPOINTS[i]; }
    o_lpts[0] = off; off += (size_t)NB * 6 * NIN;
    for (int i = 0; i < 4; ++i) { o_lpts[i + 1] = off; off += (size_t)NB * H_MLPS[i][2] * H_NPOINTS[i]; }

    // ---- workspace bump allocator
    char* wsb = (char*)d_ws;
    size_t woff = 0;
    auto walloc = [&](size_t bytes) -> void* {
        void* p = wsb + woff;
        woff = (woff + bytes + 255) & ~(size_t)255;
        return p;
    };
    float*    xyzA   = (float*)walloc((size_t)NB * NIN * 3 * 4);
    float*    xyzB   = (float*)walloc((size_t)NB * NIN * 3 * 4);
    float*    ptsA   = (float*)walloc((size_t)NB * 131072 * 4);
    float*    ptsB   = (float*)walloc((size_t)NB * 131072 * 4);
    int*      fidx   = (int*)walloc((size_t)NB * 2048 * 4);
    int*      bidx   = (int*)walloc((size_t)NB * 2048 * NS * 4);
    _Float16* Xa     = (_Float16*)walloc((size_t)33554432);
    _Float16* Xb     = (_Float16*)walloc((size_t)33554432);
    float*    Ybuf   = (float*)walloc((size_t)67108864);
    _Float16* Wh     = (_Float16*)walloc((size_t)512 * 512 * 2);
    float*    meanb  = (float*)walloc(512 * 4);
    float*    istdb  = (float*)walloc(512 * 4);

    auto blk = [](long n) { return (int)((n + 255) / 256); };

    // stage 0 init
    k_init<<<blk((long)NB * NIN), 256, 0, stream>>>(xyz_in, xyzA, ptsA,
                                                    out + o_lxyz[0], out + o_lpts[0]);

    float* cxyz = xyzA;  float* nxyz = xyzB;
    float* cpts = ptsA;  float* npts = ptsB;
    int N = NIN, C = 6, pi = 1;

    for (int st = 0; st < 4; ++st) {
        int S = H_NPOINTS[st];
        int M = NB * S * NS;

        // FPS (one workgroup per batch, dist in LDS)
        size_t smem = (size_t)(N + 2 * FPS_T) * 4;
        k_fps<<<NB, FPS_T, smem, stream>>>(cxyz, fidx, N, S);

        // new_xyz into next stage's xyz buffer; also emits l_xyz[st+1]
        k_newxyz<<<blk((long)NB * S), 256, 0, stream>>>(cxyz, fidx, nxyz,
                                                        out + o_lxyz[st + 1], N, S);

        // ball query + grouping into half X
        float r = H_RADII[st];
        k_ball<<<blk((long)NB * S), 256, 0, stream>>>(cxyz, nxyz, bidx, N, S, r * r);
        int Cin = C + 3;
        int Cpad = ((Cin + 31) / 32) * 32;
        k_group<<<blk((long)M), 256, 0, stream>>>(cxyz, cpts, nxyz, bidx, Xa, N, S, C, Cpad);

        _Float16* Xcur = Xa; _Float16* Xnxt = Xb;
        for (int l = 0; l < 3; ++l) {
            const float* W  = (const float*)d_in[pi++];
            const float* bb = (const float*)d_in[pi++];
            const float* g  = (const float*)d_in[pi++];
            const float* be = (const float*)d_in[pi++];
            int Cout = H_MLPS[st][l];

            k_w2h<<<blk((long)Cout * Cpad), 256, 0, stream>>>(W, Wh, Cout, Cin, Cpad);

            // 2x2 tile block per wave: grid = (M/32/GEMM_WAVES, Cout/32)
            dim3 gg(M / 32 / GEMM_WAVES, Cout / 32);
            k_gemm<<<gg, 32 * GEMM_WAVES, 0, stream>>>(Wh, Xcur, bb, Ybuf, Cpad, M);

            k_stats<<<Cout, 256, 0, stream>>>(Ybuf, meanb, istdb, M);

            if (l < 2) {
                k_bnrelu<<<blk((long)M), 256, 0, stream>>>(Ybuf, meanb, istdb, g, be,
                                                           Xnxt, Cout, M);
                _Float16* t = Xcur; Xcur = Xnxt; Xnxt = t;
                Cin = Cout; Cpad = Cout;   // Couts are all multiples of 32
            } else {
                k_maxpool<<<blk((long)NB * S * Cout), 256, 0, stream>>>(
                    Ybuf, meanb, istdb, g, be, out + o_lpts[st + 1], npts, S, Cout);
            }
        }

        // rotate buffers for next stage
        C = H_MLPS[st][2];
        N = S;
        float* t;
        t = cxyz; cxyz = nxyz; nxyz = t;
        t = cpts; cpts = npts; npts = t;
    }
}
